// Attention_86792699118108
// MI455X (gfx1250) — matrix operand order repaired and checked by reading, not compiled, not run on hardware
//
#include <hip/hip_runtime.h>
#include <hip/hip_bf16.h>

typedef __attribute__((ext_vector_type(16))) __bf16 v16bf;
typedef __attribute__((ext_vector_type(8)))  __bf16 v8bf;
typedef __attribute__((ext_vector_type(8)))  float  v8f;
typedef __attribute__((ext_vector_type(4)))  float  v4f;

union Frag { v16bf v; v8bf h[2]; };

#define C_DIM   768
#define NQKV    2304
#define M_TOT   50176   // 16 * 56 * 56
#define WSZ     14
#define NHEAD   12
#define HD      64
#define NWIN    256
#define NTOK    196
#define NPAD    224     // 14 tiles of 16; 7 chunks of 32

// ---------------------------------------------------------------------------
// Kernel 1: QKV projection GEMM (bf16 WMMA, fp32 accum), software-pipelined.
// A = x [50176, 768] fp32, B = qkv_w [2304, 768] fp32 (both converted to bf16
// while staging through LDS). blockIdx.x = N tile (fast) so L2 keeps the A
// tile + full weight matrix resident; x streams from HBM exactly once.
// Output window-partitioned: q/k [win][head][n][hd], v [win][head][hd][n].
// ---------------------------------------------------------------------------
__global__ __launch_bounds__(128) void qkv_gemm(
    const float* __restrict__ x, const float* __restrict__ w,
    const float* __restrict__ qb, const float* __restrict__ vb,
    __bf16* __restrict__ q_ws, __bf16* __restrict__ k_ws, __bf16* __restrict__ v_ws)
{
    __shared__ __attribute__((aligned(16))) __bf16 As[64][40]; // [m][k], 8-elem pad
    __shared__ __attribute__((aligned(16))) __bf16 Bs[64][40]; // [n][k]

    const int tidx = threadIdx.x;
    const int wave = tidx >> 5;
    const int lane = tidx & 31;
    const int lm   = lane & 15;
    const int kh   = lane >> 4;
    const int n0   = blockIdx.x * 64;   // N fast-varying -> L2 reuse of A
    const int m0   = blockIdx.y * 64;
    const int wm   = (wave >> 1) * 32;
    const int wn   = (wave & 1) * 32;

    v8f acc[2][2];
    const v8f vz = {0.f,0.f,0.f,0.f,0.f,0.f,0.f,0.f};
    acc[0][0] = vz; acc[0][1] = vz; acc[1][0] = vz; acc[1][1] = vz;

    // Prefetch registers: 64x32 tile = 512 float4 / 128 threads = 4 each.
    v4f aReg[4], bReg[4];
    #pragma unroll
    for (int p = 0; p < 4; ++p) {
        const int j = tidx + 128 * p;          // float4 index
        const int r = j >> 3, c = (j & 7) * 4; // 8 float4 per 32-wide row
        aReg[p] = *(const v4f*)&x[(size_t)(m0 + r) * C_DIM + c];
        bReg[p] = *(const v4f*)&w[(size_t)(n0 + r) * C_DIM + c];
    }

    for (int k0 = 0; k0 < C_DIM; k0 += 32) {
        __syncthreads();   // previous chunk's fragment reads done
        #pragma unroll
        for (int p = 0; p < 4; ++p) {
            const int j = tidx + 128 * p;
            const int r = j >> 3, c = (j & 7) * 4;
            #pragma unroll
            for (int e = 0; e < 4; ++e) {
                As[r][c + e] = (__bf16)aReg[p][e];
                Bs[r][c + e] = (__bf16)bReg[p][e];
            }
        }
        __syncthreads();   // stores visible
        if (k0 + 32 < C_DIM) {
            #pragma unroll
            for (int p = 0; p < 4; ++p) {
                const int j = tidx + 128 * p;
                const int r = j >> 3, c = (j & 7) * 4;
                aReg[p] = *(const v4f*)&x[(size_t)(m0 + r) * C_DIM + k0 + 32 + c];
                bReg[p] = *(const v4f*)&w[(size_t)(n0 + r) * C_DIM + k0 + 32 + c];
            }
        }

        Frag a[2], b[2];
        #pragma unroll
        for (int t = 0; t < 2; ++t) {
            // A frag: element j -> k = (j<8 ? j : j+8) + 8*kh
            a[t].h[0] = *(const v8bf*)&As[wm + t * 16 + lm][8 * kh];
            a[t].h[1] = *(const v8bf*)&As[wm + t * 16 + lm][16 + 8 * kh];
            // B frag: element j -> k = (j<8 ? j : j+8) + 8*kh
            b[t].h[0] = *(const v8bf*)&Bs[wn + t * 16 + lm][8 * kh];
            b[t].h[1] = *(const v8bf*)&Bs[wn + t * 16 + lm][16 + 8 * kh];
        }
        #pragma unroll
        for (int tm = 0; tm < 2; ++tm)
            #pragma unroll
            for (int tn = 0; tn < 2; ++tn)
                acc[tm][tn] = __builtin_amdgcn_wmma_f32_16x16x32_bf16(
                    false, a[tm].v, false, b[tn].v, (short)0, acc[tm][tn], false, false);
    }

    // Fused store: window partition + bias + q-scale, bf16 output.
    #pragma unroll
    for (int tm = 0; tm < 2; ++tm) {
        #pragma unroll
        for (int tn = 0; tn < 2; ++tn) {
            #pragma unroll
            for (int r = 0; r < 8; ++r) {
                const int m = m0 + wm + tm * 16 + r + 8 * kh;  // row in x order
                const int n = n0 + wn + tn * 16 + lm;          // col in [0,2304)
                float val = acc[tm][tn][r];
                const int bb  = m / 3136;
                const int rem = m % 3136;
                const int hh  = rem / 56;
                const int ww  = rem % 56;
                const int win = bb * 16 + (hh / WSZ) * 4 + (ww / WSZ);
                const int tok = (hh % WSZ) * WSZ + (ww % WSZ);
                const int which = n / C_DIM;
                const int col   = n % C_DIM;
                const int head  = col >> 6;
                const int hd    = col & 63;
                const size_t base = (size_t)win * NHEAD + head;
                if (which == 0) {
                    val = (val + qb[col]) * 0.125f;  // hd^-0.5 = 1/8
                    q_ws[(base * NPAD + tok) * HD + hd] = (__bf16)val;
                } else if (which == 1) {
                    k_ws[(base * NPAD + tok) * HD + hd] = (__bf16)val;
                } else {
                    val += vb[col];
                    v_ws[(base * HD + hd) * NPAD + tok] = (__bf16)val;  // V transposed
                }
            }
        }
    }
}

// ---------------------------------------------------------------------------
// Kernel 2: per-(window, head, query-tile) attention. One wave per block.
// S = q·k^T (+rel bias, mask), fp32 softmax, P·V, all via bf16 WMMA.
// ---------------------------------------------------------------------------
__global__ __launch_bounds__(32) void attn_kernel(
    const __bf16* __restrict__ q_ws, const __bf16* __restrict__ k_ws,
    const __bf16* __restrict__ v_ws, const float* __restrict__ rbt,
    __bf16* __restrict__ o_ws)
{
    __shared__ __attribute__((aligned(16))) __bf16 P[16][NPAD];

    const int lane = threadIdx.x;
    const int lm   = lane & 15;
    const int kh   = lane >> 4;
    const int qt   = blockIdx.x;   // 0..12 (13 tiles cover 196 rows)
    const int head = blockIdx.y;
    const int win  = blockIdx.z;
    const size_t base = ((size_t)win * NHEAD + head) * (size_t)(NPAD * HD);

    // Q A-fragments: row = qt*16+lm, two 32-deep chunks over hd=64.
    Frag aq[2];
    const __bf16* qrow = q_ws + base + (size_t)(qt * 16 + lm) * HD;
    #pragma unroll
    for (int c = 0; c < 2; ++c) {
        aq[c].h[0] = *(const v8bf*)&qrow[32 * c + 8 * kh];
        aq[c].h[1] = *(const v8bf*)&qrow[32 * c + 16 + 8 * kh];
    }

    // S = q · k^T over 14 key tiles.
    v8f s[14];
    const v8f vz = {0.f,0.f,0.f,0.f,0.f,0.f,0.f,0.f};
    #pragma unroll
    for (int kt = 0; kt < 14; ++kt) s[kt] = vz;

    #pragma unroll
    for (int kt = 0; kt < 14; ++kt) {
        const __bf16* krow = k_ws + base + (size_t)(kt * 16 + lm) * HD;
        #pragma unroll
        for (int c = 0; c < 2; ++c) {
            Frag bk;  // B frag: element j -> hd = 32c + (j<8?j:j+8) + 8*kh, col = key
            bk.h[0] = *(const v8bf*)&krow[32 * c + 8 * kh];
            bk.h[1] = *(const v8bf*)&krow[32 * c + 16 + 8 * kh];
            s[kt] = __builtin_amdgcn_wmma_f32_16x16x32_bf16(
                false, aq[c].v, false, bk.v, (short)0, s[kt], false, false);
        }
    }

    // Relative-position bias + mask (keys/rows >= 196 -> -1e30).
    #pragma unroll
    for (int kt = 0; kt < 14; ++kt) {
        #pragma unroll
        for (int r = 0; r < 8; ++r) {
            const int qi = qt * 16 + r + 8 * kh;
            const int kj = kt * 16 + lm;
            float v = s[kt][r];
            if (qi < NTOK && kj < NTOK) {
                const int qy = qi / WSZ, qx = qi % WSZ;
                const int ky = kj / WSZ, kx = kj % WSZ;
                const int idx = (qy - ky + WSZ - 1) * (2 * WSZ - 1) + (qx - kx + WSZ - 1);
                v += rbt[idx * NHEAD + head];
            } else {
                v = -1e30f;
            }
            s[kt][r] = v;
        }
    }

    // Row softmax: reduce over 14 regs, then 16 lanes (xor 1,2,4,8 stays in half).
    #pragma unroll
    for (int r = 0; r < 8; ++r) {
        float m = -3.4e38f;
        #pragma unroll
        for (int kt = 0; kt < 14; ++kt) m = fmaxf(m, s[kt][r]);
        #pragma unroll
        for (int off = 1; off < 16; off <<= 1) m = fmaxf(m, __shfl_xor(m, off, 32));
        float sum = 0.f;
        #pragma unroll
        for (int kt = 0; kt < 14; ++kt) {
            const float e = __expf(s[kt][r] - m);
            s[kt][r] = e;
            sum += e;
        }
        #pragma unroll
        for (int off = 1; off < 16; off <<= 1) sum += __shfl_xor(sum, off, 32);
        const float inv = 1.0f / sum;
        #pragma unroll
        for (int kt = 0; kt < 14; ++kt)
            P[r + 8 * kh][kt * 16 + lm] = (__bf16)(s[kt][r] * inv);
    }
    __syncthreads();

    // O = P · V : 7 chunks of 32 keys, 4 hd tiles of 16.
    v8f o[4];
    #pragma unroll
    for (int nt = 0; nt < 4; ++nt) o[nt] = vz;

    #pragma unroll
    for (int c = 0; c < 7; ++c) {
        Frag ap;  // A frag from LDS P: element j -> key = 32c + (j<8?j:j+8) + 8*kh
        ap.h[0] = *(const v8bf*)&P[lm][32 * c + 8 * kh];
        ap.h[1] = *(const v8bf*)&P[lm][32 * c + 16 + 8 * kh];
        #pragma unroll
        for (int nt = 0; nt < 4; ++nt) {
            const __bf16* vr = v_ws + base + (size_t)(nt * 16 + lm) * NPAD;
            Frag bv;  // B frag from V^T: element j -> key = 32c + (j<8?j:j+8) + 8*kh
            bv.h[0] = *(const v8bf*)&vr[32 * c + 8 * kh];
            bv.h[1] = *(const v8bf*)&vr[32 * c + 16 + 8 * kh];
            o[nt] = __builtin_amdgcn_wmma_f32_16x16x32_bf16(
                false, ap.v, false, bv.v, (short)0, o[nt], false, false);
        }
    }

    // Store O (bf16) as [win*196 + tok, 768], head-major columns.
    #pragma unroll
    for (int nt = 0; nt < 4; ++nt) {
        #pragma unroll
        for (int r = 0; r < 8; ++r) {
            const int qi = qt * 16 + r + 8 * kh;
            if (qi < NTOK) {
                const size_t row = (size_t)win * NTOK + qi;
                o_ws[row * C_DIM + head * HD + nt * 16 + lm] = (__bf16)o[nt][r];
            }
        }
    }
}

// ---------------------------------------------------------------------------
// Kernel 3: output projection GEMM + bias + window-reverse, fp32 output.
// Same pipelined scheme; A tile is already bf16 (16B vector loads).
// ---------------------------------------------------------------------------
__global__ __launch_bounds__(128) void proj_gemm(
    const __bf16* __restrict__ o_ws, const float* __restrict__ w,
    const float* __restrict__ pb, float* __restrict__ out)
{
    __shared__ __attribute__((aligned(16))) __bf16 As[64][40];
    __shared__ __attribute__((aligned(16))) __bf16 Bs[64][40];

    const int tidx = threadIdx.x;
    const int wave = tidx >> 5;
    const int lane = tidx & 31;
    const int lm   = lane & 15;
    const int kh   = lane >> 4;
    const int n0   = blockIdx.x * 64;   // N fast-varying
    const int m0   = blockIdx.y * 64;
    const int wm   = (wave >> 1) * 32;
    const int wn   = (wave & 1) * 32;

    v8f acc[2][2];
    const v8f vz = {0.f,0.f,0.f,0.f,0.f,0.f,0.f,0.f};
    acc[0][0] = vz; acc[0][1] = vz; acc[1][0] = vz; acc[1][1] = vz;

    // A: 2048 bf16 / 128 threads = 2 x v8bf each. B: 4 x float4 each.
    v8bf aReg[2];
    v4f  bReg[4];
    #pragma unroll
    for (int p = 0; p < 2; ++p) {
        const int j = tidx + 128 * p;          // v8bf index
        const int r = j >> 2, c = (j & 3) * 8; // 4 v8bf per 32-wide row
        aReg[p] = *(const v8bf*)&o_ws[(size_t)(m0 + r) * C_DIM + c];
    }
    #pragma unroll
    for (int p = 0; p < 4; ++p) {
        const int j = tidx + 128 * p;
        const int r = j >> 3, c = (j & 7) * 4;
        bReg[p] = *(const v4f*)&w[(size_t)(n0 + r) * C_DIM + c];
    }

    for (int k0 = 0; k0 < C_DIM; k0 += 32) {
        __syncthreads();
        #pragma unroll
        for (int p = 0; p < 2; ++p) {
            const int j = tidx + 128 * p;
            const int r = j >> 2, c = (j & 3) * 8;
            *(v8bf*)&As[r][c] = aReg[p];
        }
        #pragma unroll
        for (int p = 0; p < 4; ++p) {
            const int j = tidx + 128 * p;
            const int r = j >> 3, c = (j & 7) * 4;
            #pragma unroll
            for (int e = 0; e < 4; ++e) Bs[r][c + e] = (__bf16)bReg[p][e];
        }
        __syncthreads();
        if (k0 + 32 < C_DIM) {
            #pragma unroll
            for (int p = 0; p < 2; ++p) {
                const int j = tidx + 128 * p;
                const int r = j >> 2, c = (j & 3) * 8;
                aReg[p] = *(const v8bf*)&o_ws[(size_t)(m0 + r) * C_DIM + k0 + 32 + c];
            }
            #pragma unroll
            for (int p = 0; p < 4; ++p) {
                const int j = tidx + 128 * p;
                const int r = j >> 3, c = (j & 7) * 4;
                bReg[p] = *(const v4f*)&w[(size_t)(n0 + r) * C_DIM + k0 + 32 + c];
            }
        }

        Frag a[2], b[2];
        #pragma unroll
        for (int t = 0; t < 2; ++t) {
            a[t].h[0] = *(const v8bf*)&As[wm + t * 16 + lm][8 * kh];
            a[t].h[1] = *(const v8bf*)&As[wm + t * 16 + lm][16 + 8 * kh];
            b[t].h[0] = *(const v8bf*)&Bs[wn + t * 16 + lm][8 * kh];
            b[t].h[1] = *(const v8bf*)&Bs[wn + t * 16 + lm][16 + 8 * kh];
        }
        #pragma unroll
        for (int tm = 0; tm < 2; ++tm)
            #pragma unroll
            for (int tn = 0; tn < 2; ++tn)
                acc[tm][tn] = __builtin_amdgcn_wmma_f32_16x16x32_bf16(
                    false, a[tm].v, false, b[tn].v, (short)0, acc[tm][tn], false, false);
    }

    // Fused store: + proj bias, window reverse to [B, H*W, C], fp32.
    #pragma unroll
    for (int tm = 0; tm < 2; ++tm) {
        #pragma unroll
        for (int tn = 0; tn < 2; ++tn) {
            #pragma unroll
            for (int r = 0; r < 8; ++r) {
                const int m = m0 + wm + tm * 16 + r + 8 * kh;  // win-order row
                const int n = n0 + wn + tn * 16 + lm;
                const int win = m / NTOK;
                const int tok = m % NTOK;
                const int bb = win >> 4;
                const int wy = (win >> 2) & 3;
                const int wx = win & 3;
                const int iy = tok / WSZ, ix = tok % WSZ;
                const int hh = wy * WSZ + iy;
                const int ww = wx * WSZ + ix;
                out[((size_t)bb * 3136 + (size_t)hh * 56 + ww) * C_DIM + n] =
                    acc[tm][tn][r] + pb[n];
            }
        }
    }
}

// ---------------------------------------------------------------------------
extern "C" void kernel_launch(void* const* d_in, const int* in_sizes, int n_in,
                              void* d_out, int out_size, void* d_ws, size_t ws_size,
                              hipStream_t stream) {
    (void)in_sizes; (void)n_in; (void)out_size; (void)ws_size;
    const float* x      = (const float*)d_in[0];
    const float* qkv_w  = (const float*)d_in[1];
    const float* q_bias = (const float*)d_in[2];
    const float* v_bias = (const float*)d_in[3];
    const float* rbt    = (const float*)d_in[4];
    const float* proj_w = (const float*)d_in[5];
    const float* proj_b = (const float*)d_in[6];
    float* out = (float*)d_out;

    const size_t qkv_elems = (size_t)NWIN * NHEAD * NPAD * HD;  // 44,040,192
    __bf16* q_ws = (__bf16*)d_ws;
    __bf16* k_ws = q_ws + qkv_elems;
    __bf16* v_ws = k_ws + qkv_elems;
    __bf16* o_ws = v_ws + qkv_elems;

    qkv_gemm<<<dim3(NQKV / 64, M_TOT / 64), 128, 0, stream>>>(
        x, qkv_w, q_bias, v_bias, q_ws, k_ws, v_ws);
    attn_kernel<<<dim3(13, NHEAD, NWIN), 32, 0, stream>>>(
        q_ws, k_ws, v_ws, rbt, o_ws);
    proj_gemm<<<dim3(C_DIM / 64, M_TOT / 64), 128, 0, stream>>>(
        o_ws, proj_w, proj_b, out);
}
